// NonLinearBiasReadoutBlock_20864951124193
// MI455X (gfx1250) — compile-verified, matrix operand order checked by reading, not run on hardware
//
#include <hip/hip_runtime.h>

// MI455X / gfx1250 fused kernel for:
//   h = silu(x @ W1 / 32); h = silu(h @ W_mid / 32 + b_mid);
//   h = head_mask(h);       out = h @ W2 / 32 + b2
// N=100000, C=1024, H=4. Both big GEMMs run on v_wmma_f32_16x16x32_bf16.
// Activations live in LDS as packed bf16 pairs so A-fragments are pure
// ds_load_b128 (no per-k VALU conversion).

typedef __attribute__((ext_vector_type(8)))  float  v8f;
typedef __attribute__((ext_vector_type(16))) __bf16 v16bf;

union BF8 {
  unsigned int u[8];
  v16bf v;
};

#define INV_SQRT 0.03125f   // 1/sqrt(1024)
#define NT_TOTAL 64         // 1024 / 16 column tiles
#define KT_TOTAL 32         // 1024 / 32 k tiles
#define LDS_STRIDE 520      // u32 per activation row (512 + pad, 16B-aligned)

__device__ __forceinline__ unsigned int pack_bf16(float a, float b) {
  unsigned int ua = __float_as_uint(a);
  unsigned int ub = __float_as_uint(b);
  ua += 0x7FFFu + ((ua >> 16) & 1u);   // round-to-nearest-even
  ub += 0x7FFFu + ((ub >> 16) & 1u);
  return (ua >> 16) | (ub & 0xFFFF0000u);
}

__device__ __forceinline__ float silu(float v) {
  return v / (1.0f + __expf(-v));
}

// ---------------------------------------------------------------------------
// Weight repack: f32 [K=1024][N=1024] row-major -> bf16 B-fragment layout
// index = (((nt*32 + kt)*32 + lane)*8 + vreg) dwords, matching the ISA's
// 16-bit 32x16 B-matrix VGPR layout (lane<16: K=0..7,16..23; lane>=16: +8).
// ---------------------------------------------------------------------------
__global__ __launch_bounds__(256)
void pack_weight_kernel(const float* __restrict__ W, unsigned int* __restrict__ Wp) {
  int gid = blockIdx.x * 256 + threadIdx.x;          // (nt*32 + kt)*32 + lane
  if (gid >= NT_TOTAL * KT_TOTAL * 32) return;
  int lane = gid & 31;
  int kt   = (gid >> 5) & 31;
  int nt   = gid >> 10;
  int col   = nt * 16 + (lane & 15);
  int kbase = kt * 32 + ((lane & 16) ? 8 : 0);
  unsigned int* o = Wp + (size_t)gid * 8;
#pragma unroll
  for (int v = 0; v < 4; ++v) {
    int k0 = kbase + 2 * v;
    o[v]     = pack_bf16(W[(size_t)k0 * 1024 + col], W[(size_t)(k0 + 1) * 1024 + col]);
    int k1 = kbase + 16 + 2 * v;
    o[4 + v] = pack_bf16(W[(size_t)k1 * 1024 + col], W[(size_t)(k1 + 1) * 1024 + col]);
  }
}

// ---------------------------------------------------------------------------
// One GEMM pass: acc[rt][i] += A(LDS, packed bf16) @ B(global, packed bf16).
// Wave owns row-tiles rt=0,1 and column tiles nt0..nt0+7.
// A fragment = two ds_load_b128 per row tile; zero conversion VALU.
// ---------------------------------------------------------------------------
__device__ __forceinline__ void gemm_pass(const unsigned int* actu,
                                          const unsigned int* __restrict__ Bp,
                                          int lane, int nt0, v8f acc[2][8]) {
  const int half = (lane & 16) ? 4 : 0;   // lane>=16 owns K+8 pairs
  for (int kt = 0; kt < KT_TOTAL; ++kt) {
    BF8 af[2];
#pragma unroll
    for (int rt = 0; rt < 2; ++rt) {
      const uint4* p = (const uint4*)(actu + (size_t)(rt * 16 + (lane & 15)) * LDS_STRIDE
                                      + kt * 16 + half);
      uint4 q0 = p[0];          // u32 pairs: K {0..7} (or {8..15})
      uint4 q1 = p[2];          // u32 pairs: K {16..23} (or {24..31})
      af[rt].u[0] = q0.x; af[rt].u[1] = q0.y; af[rt].u[2] = q0.z; af[rt].u[3] = q0.w;
      af[rt].u[4] = q1.x; af[rt].u[5] = q1.y; af[rt].u[6] = q1.z; af[rt].u[7] = q1.w;
    }
#pragma unroll
    for (int i = 0; i < 8; ++i) {
      const uint4* bp =
          (const uint4*)(Bp + (size_t)((((nt0 + i) * 32 + kt) * 32 + lane) * 8));
      uint4 b0 = bp[0], b1 = bp[1];                   // two coalesced b128 loads
      BF8 bf;
      bf.u[0] = b0.x; bf.u[1] = b0.y; bf.u[2] = b0.z; bf.u[3] = b0.w;
      bf.u[4] = b1.x; bf.u[5] = b1.y; bf.u[6] = b1.z; bf.u[7] = b1.w;
      acc[0][i] = __builtin_amdgcn_wmma_f32_16x16x32_bf16(
          false, af[0].v, false, bf.v, (short)0, acc[0][i], false, false);
      acc[1][i] = __builtin_amdgcn_wmma_f32_16x16x32_bf16(
          false, af[1].v, false, bf.v, (short)0, acc[1][i], false, false);
    }
  }
}

// ---------------------------------------------------------------------------
// Fused main kernel: 32 rows per block, 8 waves; activations stay in LDS
// (packed bf16) between the two WMMA GEMMs; readout fused in registers.
// ---------------------------------------------------------------------------
__global__ __launch_bounds__(256)
void fused_mlp_kernel(const float* __restrict__ x, const int* __restrict__ heads,
                      const unsigned int* __restrict__ W1p,
                      const unsigned int* __restrict__ Wmp,
                      const float* __restrict__ b_mid, const float* __restrict__ W2,
                      const float* __restrict__ b2, float* __restrict__ out) {
  __shared__ unsigned int actu[32 * LDS_STRIDE];   // packed bf16 pairs, ~65 KB
  __shared__ float part[8 * 128];                  // per-wave partial readouts
  __shared__ int   hid[32];

  const int tid  = threadIdx.x;
  const int lane = tid & 31;
  const int wave = tid >> 5;
  const int rowbase = blockIdx.x * 32;    // N = 100000 = 3125 * 32 exactly
  const int nt0 = wave * 8;

  // ---- stage x tile: global f32 -> LDS packed bf16 (convert once) ----
  const float4* xg = (const float4*)(x + (size_t)rowbase * 1024);
  for (int idx = tid; idx < 32 * 256; idx += 256) {
    int row = idx >> 8, c4 = idx & 255;
    float4 v = xg[(size_t)row * 256 + c4];
    uint2 pk;
    pk.x = pack_bf16(v.x, v.y);
    pk.y = pack_bf16(v.z, v.w);
    *((uint2*)&actu[(size_t)row * LDS_STRIDE + c4 * 2]) = pk;
  }
  if (tid < 32) hid[tid] = heads[rowbase + tid];
  __syncthreads();

  // ---- GEMM1: x @ W1 ----
  v8f acc[2][8];
#pragma unroll
  for (int rt = 0; rt < 2; ++rt)
#pragma unroll
    for (int i = 0; i < 8; ++i) acc[rt][i] = (v8f){0, 0, 0, 0, 0, 0, 0, 0};
  gemm_pass(actu, W1p, lane, nt0, acc);
  __syncthreads();                        // everyone done reading x tile

  // ---- h1 = silu(acc/32), packed to bf16 pairs, back into LDS ----
  // C/D layout: vgpr a -> M = a + (lane<16 ? 0 : 8), N = lane%16.
  // Adjacent columns sit in adjacent lanes: shfl_xor(1) pairs them.
#pragma unroll
  for (int rt = 0; rt < 2; ++rt) {
    int rbase = rt * 16 + ((lane & 16) ? 8 : 0);
#pragma unroll
    for (int i = 0; i < 8; ++i) {
      int colp = (nt0 + i) * 8 + ((lane & 15) >> 1);   // packed u32 column
#pragma unroll
      for (int a = 0; a < 8; ++a) {
        float v  = silu(acc[rt][i][a] * INV_SQRT);
        float vp = __shfl_xor(v, 1, 32);               // partner column
        unsigned int pk = pack_bf16(v, vp);            // valid on even lanes
        if ((lane & 1) == 0) actu[(size_t)(rbase + a) * LDS_STRIDE + colp] = pk;
      }
    }
  }
  __syncthreads();

  // ---- GEMM2: h1 @ W_mid ----
#pragma unroll
  for (int rt = 0; rt < 2; ++rt)
#pragma unroll
    for (int i = 0; i < 8; ++i) acc[rt][i] = (v8f){0, 0, 0, 0, 0, 0, 0, 0};
  gemm_pass(actu, Wmp, lane, nt0, acc);

  // ---- fused epilogue: bias + silu + head mask + 1024->4 readout ----
  int hrow[2][8];
#pragma unroll
  for (int rt = 0; rt < 2; ++rt) {
    int rbase = rt * 16 + ((lane & 16) ? 8 : 0);
#pragma unroll
    for (int a = 0; a < 8; ++a) hrow[rt][a] = hid[rbase + a];
  }

  float pj[2][8][4];
#pragma unroll
  for (int rt = 0; rt < 2; ++rt)
#pragma unroll
    for (int a = 0; a < 8; ++a)
#pragma unroll
      for (int j = 0; j < 4; ++j) pj[rt][a][j] = 0.0f;

#pragma unroll
  for (int i = 0; i < 8; ++i) {
    int nt = nt0 + i;
    int c  = nt * 16 + (lane & 15);
    float  bm = b_mid[c];
    float4 w2 = ((const float4*)W2)[c];   // W2 is [1024][4] row-major
    int headc = nt >> 4;                  // head id is uniform per 16-col tile
#pragma unroll
    for (int rt = 0; rt < 2; ++rt) {
#pragma unroll
      for (int a = 0; a < 8; ++a) {
        float v = silu(acc[rt][i][a] * INV_SQRT + bm);
        float hv = (hrow[rt][a] == headc) ? v : 0.0f;
        pj[rt][a][0] += hv * w2.x;
        pj[rt][a][1] += hv * w2.y;
        pj[rt][a][2] += hv * w2.z;
        pj[rt][a][3] += hv * w2.w;
      }
    }
  }

  // butterfly-reduce the 16 columns held within each 16-lane half (wave32)
#pragma unroll
  for (int rt = 0; rt < 2; ++rt) {
    int rbase = rt * 16 + ((lane & 16) ? 8 : 0);
#pragma unroll
    for (int a = 0; a < 8; ++a) {
#pragma unroll
      for (int j = 0; j < 4; ++j) {
        float s = pj[rt][a][j];
        s += __shfl_xor(s, 1, 32);
        s += __shfl_xor(s, 2, 32);
        s += __shfl_xor(s, 4, 32);
        s += __shfl_xor(s, 8, 32);
        if ((lane & 15) == 0) part[wave * 128 + (rbase + a) * 4 + j] = s;
      }
    }
  }
  __syncthreads();

  // deterministic cross-wave combine + final scale/bias
  if (tid < 128) {
    int row = tid >> 2, j = tid & 3;
    float s = 0.0f;
#pragma unroll
    for (int w = 0; w < 8; ++w) s += part[w * 128 + tid];
    out[(size_t)(rowbase + row) * 4 + j] = s * INV_SQRT + b2[j];
  }
}

// ---------------------------------------------------------------------------
extern "C" void kernel_launch(void* const* d_in, const int* in_sizes, int n_in,
                              void* d_out, int out_size, void* d_ws, size_t ws_size,
                              hipStream_t stream) {
  const float* x     = (const float*)d_in[0];
  const int*   heads = (const int*)d_in[1];
  const float* W1    = (const float*)d_in[2];
  const float* W_mid = (const float*)d_in[3];
  const float* b_mid = (const float*)d_in[4];
  const float* W2    = (const float*)d_in[5];
  const float* b2    = (const float*)d_in[6];
  float*       out   = (float*)d_out;

  // workspace: 2 MB packed bf16 per weight matrix (524288 dwords each)
  unsigned int* W1p = (unsigned int*)d_ws;
  unsigned int* Wmp = W1p + 524288;

  const int pack_threads = NT_TOTAL * KT_TOTAL * 32;   // 65536
  pack_weight_kernel<<<pack_threads / 256, 256, 0, stream>>>(W1, W1p);
  pack_weight_kernel<<<pack_threads / 256, 256, 0, stream>>>(W_mid, Wmp);

  const int n_blocks = 100000 / 32;                    // 3125, exact
  fused_mlp_kernel<<<n_blocks, 256, 0, stream>>>(x, heads, W1p, Wmp,
                                                 b_mid, W2, b2, out);
}
